// MultiHeadAttention_33741263078128
// MI455X (gfx1250) — compile-verified, matrix operand order checked by reading, not run on hardware
//
#include <hip/hip_runtime.h>
#include <hip/hip_bf16.h>
#include <math.h>

// ---------------------------------------------------------------------------
// MHA forward for (b=2, l=2048, d=1024, h=16, dh=64), fp32 in/out.
// All GEMMs (QKV proj, QK^T, AV, out proj) run on v_wmma_f32_16x16x32_bf16.
// GEMM: ping-pong register prefetch. Attention: double-buffered LDS K/V tiles.
// ---------------------------------------------------------------------------

typedef __attribute__((ext_vector_type(16))) __bf16          v16bf;
typedef __attribute__((ext_vector_type(8)))  float           v8f;
typedef __attribute__((ext_vector_type(8)))  unsigned short  us8;

union Frag {
    v16bf bf;
    us8   h[2];
};

static constexpr int Bc = 2;
static constexpr int Lc = 2048;
static constexpr int Dc = 1024;
static constexpr int Hc = 16;
static constexpr int DHc = 64;

__device__ __forceinline__ unsigned short f2u_bf16(float f) {
    unsigned int u = __float_as_uint(f);
    unsigned int r = u + 0x7FFFu + ((u >> 16) & 1u);   // round-to-nearest-even
    return (unsigned short)(r >> 16);
}

__device__ __forceinline__ v8f wmma_bf16(const Frag& a, const Frag& b, v8f c) {
    return __builtin_amdgcn_wmma_f32_16x16x32_bf16(false, a.bf, false, b.bf,
                                                   (short)0, c, false, false);
}

// ---------------------------------------------------------------------------
// fp32 -> bf16 elementwise
// ---------------------------------------------------------------------------
__global__ void k_cvt_bf16(const float* __restrict__ src,
                           unsigned short* __restrict__ dst, int n) {
    int i = blockIdx.x * blockDim.x + threadIdx.x;
    if (i < n) dst[i] = f2u_bf16(src[i]);
}

// ---------------------------------------------------------------------------
// Wt[n][k] = bf16(W[k][n])   (D x D)
// ---------------------------------------------------------------------------
__global__ void k_transpose_cvt(const float* __restrict__ W,
                                unsigned short* __restrict__ Wt) {
    int idx = blockIdx.x * blockDim.x + threadIdx.x;
    if (idx >= Dc * Dc) return;
    int k = idx / Dc;
    int n = idx % Dc;
    Wt[(size_t)n * Dc + k] = f2u_bf16(W[(size_t)k * Dc + n]);
}

// ---------------------------------------------------------------------------
// Tiled GEMM: C[L x D] = Xb[L x D] @ W (W transposed, bf16).
// 128 threads = 4 waves; block tile 128x64; wave tile 32x64 (2 A-frags reuse
// 4 B-frags). K-loop unrolled x2 with ping-pong fragment sets so the loads
// for the next 32-K slab issue before the WMMAs on the current one.
// mode 0: Q -> 1/sqrt(D) scale + RoPE -> bf16 [b][h][l][dh]
// mode 1: K -> RoPE -> bf16 [b][h][l][dh]
// mode 2: V -> bf16 transposed [b][h][dh][l]
// mode 3: out projection -> fp32 [b][l][d]
// ---------------------------------------------------------------------------
__global__ void __launch_bounds__(128)
k_gemm128(const unsigned short* __restrict__ Xb,
          const unsigned short* __restrict__ Wt,
          unsigned short* __restrict__ out16,
          float* __restrict__ out32,
          int mode)
{
    const int lane = threadIdx.x & 31;
    const int w    = threadIdx.x >> 5;
    const int half = lane >> 4;
    const int nl   = lane & 15;
    const int z    = blockIdx.z;
    const int r0   = blockIdx.x * 128 + w * 32;
    const int c0   = blockIdx.y * 64;

    const unsigned short* xrow0 = Xb + ((size_t)z * Lc + (r0 + nl)) * Dc;
    const unsigned short* xrow1 = xrow0 + (size_t)16 * Dc;

    const v8f zero8 = {0.f, 0.f, 0.f, 0.f, 0.f, 0.f, 0.f, 0.f};
    v8f acc[2][4];
#pragma unroll
    for (int mt = 0; mt < 2; ++mt)
#pragma unroll
        for (int t = 0; t < 4; ++t) acc[mt][t] = zero8;

    Frag A[2][2];     // [set][row subtile]
    Frag Bf[2][4];    // [set][col tile]

    auto LOAD = [&](int s, int kk) {
        A[s][0].h[0] = *(const us8*)(xrow0 + kk + 8 * half);
        A[s][0].h[1] = *(const us8*)(xrow0 + kk + 16 + 8 * half);
        A[s][1].h[0] = *(const us8*)(xrow1 + kk + 8 * half);
        A[s][1].h[1] = *(const us8*)(xrow1 + kk + 16 + 8 * half);
#pragma unroll
        for (int t = 0; t < 4; ++t) {
            const unsigned short* wr = Wt + (size_t)(c0 + t * 16 + nl) * Dc + kk;
            Bf[s][t].h[0] = *(const us8*)(wr + 8 * half);
            Bf[s][t].h[1] = *(const us8*)(wr + 16 + 8 * half);
        }
    };
    auto MMA = [&](int s) {
#pragma unroll
        for (int mt = 0; mt < 2; ++mt)
#pragma unroll
            for (int t = 0; t < 4; ++t)
                acc[mt][t] = wmma_bf16(A[s][mt], Bf[s][t], acc[mt][t]);
    };

    LOAD(0, 0);
#pragma unroll 1
    for (int i = 0; i < Dc / 64; ++i) {
        const int kk = i * 64;
        LOAD(1, kk + 32);
        MMA(0);
        if (i + 1 < Dc / 64) LOAD(0, kk + 64);
        MMA(1);
    }

    const float qscale = 0.03125f;               // 1/sqrt(1024)
#pragma unroll
    for (int mt = 0; mt < 2; ++mt) {
#pragma unroll
        for (int t = 0; t < 4; ++t) {
#pragma unroll
            for (int r = 0; r < 8; ++r) {
                const int M   = r + 8 * half;
                const int row = r0 + mt * 16 + M;
                const int col = c0 + t * 16 + nl;
                float v = acc[mt][t][r];
                if (mode <= 1) {
                    float partner = __shfl_xor(v, 1, 32);
                    const int e = col & 63;
                    const int p = e >> 1;
                    float inv_freq = __expf(-0.2878231366f * (float)p);
                    float ang = (float)row * inv_freq;
                    float sn, cs;
                    __sincosf(ang, &sn, &cs);
                    float o = (e & 1) ? (v * cs + partner * sn)
                                      : (v * cs - partner * sn);
                    if (mode == 0) o *= qscale;
                    const int hh = col >> 6;
                    out16[(((size_t)z * Hc + hh) * Lc + row) * DHc + e] =
                        f2u_bf16(o);
                } else if (mode == 2) {
                    const int hh = col >> 6;
                    const int e  = col & 63;
                    out16[(((size_t)z * Hc + hh) * DHc + e) * Lc + row] =
                        f2u_bf16(v);
                } else {
                    out32[((size_t)z * Lc + row) * Dc + col] = v;
                }
            }
        }
    }
}

// ---------------------------------------------------------------------------
// Flash attention per (b,h, 128-q-row block). 256 threads = 8 waves, each
// wave owns 16 q rows. K (32x64) and V^T (64x32) tiles staged cooperatively
// into double-buffered LDS; global loads for tile j+1 issue before compute
// on tile j.
// ---------------------------------------------------------------------------
__global__ void __launch_bounds__(256)
k_attn(const unsigned short* __restrict__ Qb,   // [b][h][l][dh]
       const unsigned short* __restrict__ Kb,   // [b][h][l][dh]
       const unsigned short* __restrict__ Vt,   // [b][h][dh][l]
       unsigned short* __restrict__ Yb)         // [b][l][d]
{
    __shared__ unsigned short Kt[2][32][64];    // [buf][j-row][dh]
    __shared__ unsigned short Vb[2][64][32];    // [buf][dh-row][j]
    __shared__ unsigned short Pl[8][16][32];    // per-wave P tile (16x32)

    const int tid  = threadIdx.x;
    const int lane = tid & 31;
    const int w    = tid >> 5;
    const int half = lane >> 4;
    const int nl   = lane & 15;
    const int bh   = blockIdx.y;
    const int z    = bh >> 4;
    const int hh   = bh & 15;
    const int qr0  = blockIdx.x * 128 + w * 16;

    const unsigned short* Qp = Qb + (size_t)bh * Lc * DHc;
    const unsigned short* Kp = Kb + (size_t)bh * Lc * DHc;
    const unsigned short* Vp = Vt + (size_t)bh * DHc * Lc;

    // cooperative tile copy mapping: 256 threads, one 16B chunk each
    const int krow = tid >> 3;                 // 0..31
    const int kch  = (tid & 7) * 8;            // chunk within 64-elem row
    const int vrow = tid >> 2;                 // 0..63
    const int vch  = (tid & 3) * 8;            // chunk within 32-elem row

    us8 kreg, vreg;
    auto GLOAD = [&](int j0) {
        kreg = *(const us8*)(Kp + (size_t)(j0 + krow) * DHc + kch);
        vreg = *(const us8*)(Vp + (size_t)vrow * Lc + j0 + vch);
    };
    auto LSTORE = [&](int buf) {
        *(us8*)&Kt[buf][krow][kch] = kreg;
        *(us8*)&Vb[buf][vrow][vch] = vreg;
    };

    // Q tile: 16 rows x 64 K -> two 16x32 A fragments, resident all kernel
    Frag aq[2];
    {
        const unsigned short* qrow = Qp + (size_t)(qr0 + nl) * DHc;
#pragma unroll
        for (int f = 0; f < 2; ++f) {
            aq[f].h[0] = *(const us8*)(qrow + 32 * f + 8 * half);
            aq[f].h[1] = *(const us8*)(qrow + 32 * f + 16 + 8 * half);
        }
    }

    const v8f zero8 = {0.f, 0.f, 0.f, 0.f, 0.f, 0.f, 0.f, 0.f};
    v8f   O[4];
    float m_r[8], l_r[8];
#pragma unroll
    for (int t = 0; t < 4; ++t) O[t] = zero8;
#pragma unroll
    for (int r = 0; r < 8; ++r) { m_r[r] = -1e30f; l_r[r] = 0.f; }

    // preload first tile
    GLOAD(0);
    LSTORE(0);
    __syncthreads();

    int cur = 0;
#pragma unroll 1
    for (int j0 = 0; j0 < Lc; j0 += 32) {
        const int  nxt       = cur ^ 1;
        const bool have_next = (j0 + 32 < Lc);
        if (have_next) GLOAD(j0 + 32);         // overlap with all compute below

        // ---- S = Q @ K^T (K fragments from LDS) ----
        v8f S[2];
#pragma unroll
        for (int js = 0; js < 2; ++js) {
            const unsigned short* kr = &Kt[cur][js * 16 + nl][0];
            Frag bk0, bk1;
            bk0.h[0] = *(const us8*)(kr + 8 * half);
            bk0.h[1] = *(const us8*)(kr + 16 + 8 * half);
            bk1.h[0] = *(const us8*)(kr + 32 + 8 * half);
            bk1.h[1] = *(const us8*)(kr + 48 + 8 * half);
            S[js] = wmma_bf16(aq[0], bk0, zero8);
            S[js] = wmma_bf16(aq[1], bk1, S[js]);
        }

        // ---- online softmax in C-register layout ----
        float alpha[8];
#pragma unroll
        for (int r = 0; r < 8; ++r) {
            float mx = fmaxf(S[0][r], S[1][r]);
            mx = fmaxf(mx, __shfl_xor(mx, 1, 32));
            mx = fmaxf(mx, __shfl_xor(mx, 2, 32));
            mx = fmaxf(mx, __shfl_xor(mx, 4, 32));
            mx = fmaxf(mx, __shfl_xor(mx, 8, 32));
            float mn = fmaxf(m_r[r], mx);
            alpha[r] = __expf(m_r[r] - mn);
            m_r[r]   = mn;
            float p0 = __expf(S[0][r] - mn);
            float p1 = __expf(S[1][r] - mn);
            Pl[w][r + 8 * half][nl]      = f2u_bf16(p0);
            Pl[w][r + 8 * half][16 + nl] = f2u_bf16(p1);
            float ps = p0 + p1;
            ps += __shfl_xor(ps, 1, 32);
            ps += __shfl_xor(ps, 2, 32);
            ps += __shfl_xor(ps, 4, 32);
            ps += __shfl_xor(ps, 8, 32);
            l_r[r] = l_r[r] * alpha[r] + ps;
        }
#pragma unroll
        for (int t = 0; t < 4; ++t)
#pragma unroll
            for (int r = 0; r < 8; ++r) O[t][r] *= alpha[r];

        __syncthreads();   // P visible; K-frag reads of `cur` complete

        // ---- O += P(16x32) @ V(32x64) (V fragments from LDS) ----
        Frag ap;
        ap.h[0] = *(const us8*)(&Pl[w][nl][8 * half]);
        ap.h[1] = *(const us8*)(&Pl[w][nl][16 + 8 * half]);
#pragma unroll
        for (int t = 0; t < 4; ++t) {
            const unsigned short* vr = &Vb[cur][t * 16 + nl][0];
            Frag bv;
            bv.h[0] = *(const us8*)(vr + 8 * half);
            bv.h[1] = *(const us8*)(vr + 16 + 8 * half);
            O[t] = wmma_bf16(ap, bv, O[t]);
        }

        if (have_next) LSTORE(nxt);            // consume prefetched regs
        __syncthreads();                       // next tile visible; P reads done
        cur = nxt;
    }

    // ---- finalize: O / l, interleave heads, bf16 ----
#pragma unroll
    for (int t = 0; t < 4; ++t) {
#pragma unroll
        for (int r = 0; r < 8; ++r) {
            const int M = r + 8 * half;
            float v = O[t][r] / l_r[r];
            Yb[((size_t)z * Lc + qr0 + M) * Dc + hh * DHc + t * 16 + nl] =
                f2u_bf16(v);
        }
    }
}

// ---------------------------------------------------------------------------
// Host-side launch
// ---------------------------------------------------------------------------
extern "C" void kernel_launch(void* const* d_in, const int* in_sizes, int n_in,
                              void* d_out, int out_size, void* d_ws, size_t ws_size,
                              hipStream_t stream) {
    (void)in_sizes; (void)n_in; (void)out_size; (void)ws_size;

    const float* x  = (const float*)d_in[0];
    const float* Wq = (const float*)d_in[1];
    const float* Wk = (const float*)d_in[2];
    const float* Wv = (const float*)d_in[3];
    const float* Wo = (const float*)d_in[4];
    float* out = (float*)d_out;

    char* ws = (char*)d_ws;
    size_t off = 0;
    auto alloc = [&](size_t bytes) -> char* {
        char* p = ws + off;
        off += (bytes + 255) & ~(size_t)255;
        return p;
    };

    unsigned short* xb  = (unsigned short*)alloc((size_t)Bc * Lc * Dc * 2);
    unsigned short* wt0 = (unsigned short*)alloc((size_t)Dc * Dc * 2);
    unsigned short* wt1 = (unsigned short*)alloc((size_t)Dc * Dc * 2);
    unsigned short* wt2 = (unsigned short*)alloc((size_t)Dc * Dc * 2);
    unsigned short* wt3 = (unsigned short*)alloc((size_t)Dc * Dc * 2);
    unsigned short* qb  = (unsigned short*)alloc((size_t)Bc * Hc * Lc * DHc * 2);
    unsigned short* kb  = (unsigned short*)alloc((size_t)Bc * Hc * Lc * DHc * 2);
    unsigned short* vtb = (unsigned short*)alloc((size_t)Bc * Hc * DHc * Lc * 2);
    unsigned short* yb  = (unsigned short*)alloc((size_t)Bc * Lc * Dc * 2);

    // stage 0: bf16 conversions (weights transposed)
    {
        int n = Bc * Lc * Dc;
        k_cvt_bf16<<<dim3((n + 255) / 256), dim3(256), 0, stream>>>(x, xb, n);
        int nw = Dc * Dc;
        dim3 g((nw + 255) / 256), b(256);
        k_transpose_cvt<<<g, b, 0, stream>>>(Wq, wt0);
        k_transpose_cvt<<<g, b, 0, stream>>>(Wk, wt1);
        k_transpose_cvt<<<g, b, 0, stream>>>(Wv, wt2);
        k_transpose_cvt<<<g, b, 0, stream>>>(Wo, wt3);
    }

    // stage 1: Q/K/V projections (+scale, RoPE, V transpose)
    dim3 ggrid(Lc / 128, Dc / 64, Bc), gblk(128);
    k_gemm128<<<ggrid, gblk, 0, stream>>>(xb, wt0, qb,  nullptr, 0);
    k_gemm128<<<ggrid, gblk, 0, stream>>>(xb, wt1, kb,  nullptr, 1);
    k_gemm128<<<ggrid, gblk, 0, stream>>>(xb, wt2, vtb, nullptr, 2);

    // stage 2: flash attention
    k_attn<<<dim3(Lc / 128, Bc * Hc), dim3(256), 0, stream>>>(qb, kb, vtb, yb);

    // stage 3: output projection (fp32 out)
    k_gemm128<<<ggrid, gblk, 0, stream>>>(yb, wt3, nullptr, out, 3);
}
